// DeeperMHA_19662360281286
// MI455X (gfx1250) — compile-verified, matrix operand order checked by reading, not run on hardware
//
#include <hip/hip_runtime.h>
#include <hip/hip_fp16.h>

typedef _Float16 f16;
typedef __attribute__((ext_vector_type(16))) _Float16 v16h;
typedef __attribute__((ext_vector_type(8)))  float    v8f;
typedef __attribute__((ext_vector_type(4)))  unsigned int u32x4;
typedef __attribute__((ext_vector_type(8)))  int i32x8;
typedef __attribute__((ext_vector_type(4)))  int i32x4;

#define DEV_INLINE __device__ __forceinline__

namespace dmha {

constexpr int Bb = 4, Hh = 8, S = 1024, D = 512, Dh = 64;
constexpr int BH = Bb * Hh;          // 32
constexpr int BS = Bb * S;           // 4096
constexpr int D3 = 3 * D;            // 1536
constexpr float SCALE = 0.125f;      // 1/sqrt(64)
constexpr float GN_EPS = 1e-5f;

// ---------------------------------------------------------------------------
// WMMA helpers (gfx1250 wave32, 16x16x32 f16 -> f32). All GEMMs arranged NT:
// contraction dim contiguous for both operands -> two 16B runs per lane.
// ---------------------------------------------------------------------------
DEV_INLINE v16h load_frag(const f16* __restrict__ base, int ld, int k0) {
  const int lane = threadIdx.x & 31;
  const int r = lane & 15, g = lane >> 4;
  const f16* p = base + (long)r * ld + k0 + g * 8;
  v16h f;
#pragma unroll
  for (int e = 0; e < 8; ++e) f[e] = p[e];
#pragma unroll
  for (int e = 0; e < 8; ++e) f[8 + e] = p[16 + e];
  return f;
}

DEV_INLINE v8f wmma16(const v16h& a, const v16h& b, const v8f& c) {
  return __builtin_amdgcn_wmma_f32_16x16x32_f16(false, a, false, b, (short)0, c,
                                                false, false);
}

DEV_INLINE int acc_row(int r) { return r + 8 * ((threadIdx.x & 31) >> 4); }
DEV_INLINE int acc_col() { return threadIdx.x & 15; }

// ---------------------------------------------------------------------------
// Tensor Data Mover: 2D tile (rows x cols of 2-byte elems) global -> LDS.
// D# packing per CDNA5 ISA ch.8 (group0: flags/lds/global/type, group1: dims).
// This toolchain exposes the 6-arg builtin (extra descriptor group + cpol).
// ---------------------------------------------------------------------------
#if __has_builtin(__builtin_amdgcn_tensor_load_to_lds)
#define DMHA_HAS_TDM 1
#else
#define DMHA_HAS_TDM 0
#endif

DEV_INLINE unsigned lds_off(const void* p) {
  // generic shared-aperture address: low 32 bits are the LDS byte address
  return (unsigned)(unsigned long long)(uintptr_t)p;
}

#if DMHA_HAS_TDM
DEV_INLINE void tdm_load_2d(unsigned lds_addr, const f16* gaddr, int cols,
                            int rows, int row_stride_elems) {
  unsigned long long ga = (unsigned long long)(uintptr_t)gaddr;
  u32x4 g0;
  g0[0] = 1u;                                    // count=1, no gather/restore
  g0[1] = lds_addr;                              // bits 63:32
  g0[2] = (unsigned)(ga & 0xffffffffu);          // global_addr lo
  g0[3] = (unsigned)((ga >> 32) & 0x01ffffffu) | (2u << 30);  // addr hi | type=2
  i32x8 g1;
  g1[0] = 1 << 16;                               // data_size = 1 (2 bytes)
  g1[1] = (cols & 0xffff) << 16;                 // tensor_dim0[15:0]
  g1[2] = ((cols >> 16) & 0xffff) | ((rows & 0xffff) << 16);  // dim0 hi|dim1 lo
  g1[3] = ((rows >> 16) & 0xffff) | ((cols & 0xffff) << 16);  // dim1 hi|tile_d0
  g1[4] = rows & 0xffff;                         // tile_dim1 (tile_dim2 = 0)
  g1[5] = row_stride_elems;                      // tensor_dim0_stride lo
  g1[6] = 0;
  g1[7] = 0;
  i32x4 gz4 = {0, 0, 0, 0};
  i32x8 gz8 = {0, 0, 0, 0, 0, 0, 0, 0};
  __builtin_amdgcn_tensor_load_to_lds(g0, g1, gz4, gz4, gz8, 0);
}
#endif

// ---------------------------------------------------------------------------
// 0) f32 -> f16 pack
// ---------------------------------------------------------------------------
__global__ void cvt_f32_f16(const float* __restrict__ in, f16* __restrict__ out,
                            int n) {
  int i = blockIdx.x * blockDim.x + threadIdx.x;
  if (i < n) out[i] = (f16)in[i];
}

// ---------------------------------------------------------------------------
// 1) QKV projection: [4096,512] x [1536,512]^T + bias -> q16,k16 [B,H,S,Dh],
//    v stored transposed [B,H,Dh,S] for the NT A@v GEMM.
// ---------------------------------------------------------------------------
__global__ __launch_bounds__(256) void qkv_gemm(
    const f16* __restrict__ x16, const f16* __restrict__ wp16,
    const float* __restrict__ pbias, f16* __restrict__ q16,
    f16* __restrict__ k16, f16* __restrict__ vT16) {
  const int wave = threadIdx.x >> 5;
  const int mt = blockIdx.x;             // 0..255
  const int nt = blockIdx.y * 8 + wave;  // 0..95
  const f16* abase = x16 + (long)mt * 16 * D;
  const f16* bbase = wp16 + (long)nt * 16 * D;
  v8f acc = {};
  for (int k0 = 0; k0 < D; k0 += 32) {
    v16h a = load_frag(abase, D, k0);
    v16h b = load_frag(bbase, D, k0);
    acc = wmma16(a, b, acc);
  }
  const int n = nt * 16 + acc_col();
  const int three = n / D;
  const int rem = n % D;
  const int h = rem >> 6, d = rem & 63;
  const float bias = pbias[n];
#pragma unroll
  for (int r = 0; r < 8; ++r) {
    const int row = mt * 16 + acc_row(r);  // b*S + s
    const int bidx = row >> 10, s = row & 1023;
    const f16 hv = (f16)(acc[r] + bias);
    if (three == 0)
      q16[(((long)(bidx * Hh + h)) * S + s) * Dh + d] = hv;
    else if (three == 1)
      k16[(((long)(bidx * Hh + h)) * S + s) * Dh + d] = hv;
    else
      vT16[(((long)(bidx * Hh + h)) * Dh + d) * S + s] = hv;
  }
}

// ---------------------------------------------------------------------------
// 2) KK pass 1: M[j] = max_k scale*(k_j . k_k)  (int-punned atomicMax; M >= 0)
// ---------------------------------------------------------------------------
__global__ __launch_bounds__(256) void kk_rowmax(const f16* __restrict__ k16,
                                                 float* __restrict__ Mrow) {
  const int wave = threadIdx.x >> 5;
  const int jt = blockIdx.x;
  const int kt = blockIdx.y * 8 + wave;
  const int bh = blockIdx.z;
  const f16* abase = k16 + ((long)bh * S + jt * 16) * Dh;
  const f16* bbase = k16 + ((long)bh * S + kt * 16) * Dh;
  v8f acc = {};
  for (int k0 = 0; k0 < Dh; k0 += 32) {
    v16h a = load_frag(abase, Dh, k0);
    v16h b = load_frag(bbase, Dh, k0);
    acc = wmma16(a, b, acc);
  }
  float* Mb = Mrow + (long)bh * S;
#pragma unroll
  for (int r = 0; r < 8; ++r) {
    float v = acc[r] * SCALE;
#pragma unroll
    for (int o = 1; o < 16; o <<= 1) v = fmaxf(v, __shfl_xor(v, o, 32));
    if (acc_col() == 0)
      atomicMax((int*)&Mb[jt * 16 + acc_row(r)], __float_as_int(v));
  }
}

// ---------------------------------------------------------------------------
// 3) QK GEMM: QK[bh][i][j] = scale * q_i . k_j   (stored unmasked, f32)
// ---------------------------------------------------------------------------
__global__ __launch_bounds__(256) void qk_gemm(const f16* __restrict__ q16,
                                               const f16* __restrict__ k16,
                                               float* __restrict__ QK) {
  const int wave = threadIdx.x >> 5;
  const int it = blockIdx.x;
  const int jt = blockIdx.y * 8 + wave;
  const int bh = blockIdx.z;
  const f16* abase = q16 + ((long)bh * S + it * 16) * Dh;
  const f16* bbase = k16 + ((long)bh * S + jt * 16) * Dh;
  v8f acc = {};
  for (int k0 = 0; k0 < Dh; k0 += 32) {
    v16h a = load_frag(abase, Dh, k0);
    v16h b = load_frag(bbase, Dh, k0);
    acc = wmma16(a, b, acc);
  }
  float* out = QK + (long)bh * S * S + (long)(it * 16) * S + jt * 16;
#pragma unroll
  for (int r = 0; r < 8; ++r)
    out[(long)acc_row(r) * S + acc_col()] = acc[r] * SCALE;
}

// ---------------------------------------------------------------------------
// 4) C[i] = max_{j<=i} (QK[i,j] + max(M[j],0))
// ---------------------------------------------------------------------------
__global__ __launch_bounds__(256) void row_c(const float* __restrict__ QK,
                                             const float* __restrict__ Mrow,
                                             float* __restrict__ Crow) {
  const int row = blockIdx.x;  // bh*S + i
  const int bh = row >> 10, i = row & 1023;
  const float* qk = QK + (long)bh * S * S + (long)i * S;
  const float* Mv = Mrow + (long)bh * S;
  float m = -3.4e38f;
  for (int j = threadIdx.x; j <= i; j += blockDim.x)
    m = fmaxf(m, qk[j] + fmaxf(Mv[j], 0.f));
  __shared__ float red[256];
  red[threadIdx.x] = m;
  __syncthreads();
  for (int s = 128; s > 0; s >>= 1) {
    if ((int)threadIdx.x < s)
      red[threadIdx.x] = fmaxf(red[threadIdx.x], red[threadIdx.x + s]);
    __syncthreads();
  }
  if (threadIdx.x == 0) Crow[row] = red[0];
}

// ---------------------------------------------------------------------------
// 5) E1[i,j] = (j<=i) ? exp(QK - C[i] + M[j]) : 0   (f16)
// ---------------------------------------------------------------------------
__global__ __launch_bounds__(256) void e1_kernel(const float* __restrict__ QK,
                                                 const float* __restrict__ Crow,
                                                 const float* __restrict__ Mrow,
                                                 f16* __restrict__ E1) {
  const long idx = (long)blockIdx.x * blockDim.x + threadIdx.x;
  const int bh = (int)(idx >> 20);
  const int i = (int)(idx >> 10) & 1023;
  const int j = (int)idx & 1023;
  float v = 0.f;
  if (j <= i) v = __expf(QK[idx] - Crow[bh * S + i] + Mrow[bh * S + j]);
  E1[idx] = (f16)v;
}

// ---------------------------------------------------------------------------
// 6) KK pass 2: E2T[k][j] = exp(scale*(k_j.k_k) - M[j])  (stored transposed)
// ---------------------------------------------------------------------------
__global__ __launch_bounds__(256) void kk_e2(const f16* __restrict__ k16,
                                             const float* __restrict__ Mrow,
                                             f16* __restrict__ E2T) {
  const int wave = threadIdx.x >> 5;
  const int jt = blockIdx.x;
  const int kt = blockIdx.y * 8 + wave;
  const int bh = blockIdx.z;
  const f16* abase = k16 + ((long)bh * S + jt * 16) * Dh;
  const f16* bbase = k16 + ((long)bh * S + kt * 16) * Dh;
  v8f acc = {};
  for (int k0 = 0; k0 < Dh; k0 += 32) {
    v16h a = load_frag(abase, Dh, k0);
    v16h b = load_frag(bbase, Dh, k0);
    acc = wmma16(a, b, acc);
  }
  const float* Mb = Mrow + (long)bh * S;
  f16* base = E2T + (long)bh * S * S;
#pragma unroll
  for (int r = 0; r < 8; ++r) {
    const int j = jt * 16 + acc_row(r);
    const int kg = kt * 16 + acc_col();
    base[(long)kg * S + j] = (f16)__expf(acc[r] * SCALE - Mb[j]);
  }
}

// ---------------------------------------------------------------------------
// 7) Big einsum A2 = E1 @ E2 (NT vs E2T), block-cooperative, TDM-staged LDS,
//    double-buffered; fused epilogue A = 2*exp(QK-C) - A2 with tril mask.
//    Block = 256 threads (8 waves) -> 128x64 macro-tile; per K=32 step one
//    wave DMAs an 8KB E1 panel + 4KB E2T panel (TENSORcnt), 8 waves run
//    4 WMMAs each off LDS: ~43 FLOP/byte of staged traffic.
// ---------------------------------------------------------------------------
constexpr int A2R = 128;  // macro rows (i)
constexpr int A2C = 64;   // macro cols (k)

__global__ __launch_bounds__(256) void a2_gemm(const f16* __restrict__ E1,
                                               const f16* __restrict__ E2T,
                                               const float* __restrict__ QK,
                                               const float* __restrict__ Crow,
                                               f16* __restrict__ A16) {
  __shared__ f16 sA[2][A2R * 32];
  __shared__ f16 sB[2][A2C * 32];
  const int im = blockIdx.x;   // 0..7
  const int km = blockIdx.y;   // 0..15
  const int bh = blockIdx.z;
  const int wave = threadIdx.x >> 5;
  const int i0 = im * A2R, c0 = km * A2C;
  f16* outb = A16 + (long)bh * S * S;

  if (c0 > i0 + (A2R - 1)) {  // uniform: fully above diagonal -> zeros
    for (int t = threadIdx.x; t < A2R * A2C; t += 256) {
      const int r = t >> 6, c = t & 63;
      outb[(long)(i0 + r) * S + c0 + c] = (f16)0.f;
    }
    return;
  }

  const f16* aG = E1 + (long)bh * S * S + (long)i0 * S;
  const f16* bG = E2T + (long)bh * S * S + (long)c0 * S;
  const int nsteps = (i0 + A2R) >> 5;  // causal clip (E1 zeros cover ragged)

  // stage step 0 into buffer 0
#if DMHA_HAS_TDM
  if (threadIdx.x < 32) {
    tdm_load_2d(lds_off(&sA[0][0]), aG, 32, A2R, S);
    tdm_load_2d(lds_off(&sB[0][0]), bG, 32, A2C, S);
  }
#else
  for (int t = threadIdx.x; t < A2R * 32; t += 256)
    sA[0][t] = aG[(long)(t >> 5) * S + (t & 31)];
  for (int t = threadIdx.x; t < A2C * 32; t += 256)
    sB[0][t] = bG[(long)(t >> 5) * S + (t & 31)];
#endif

  v8f acc[4] = {{}, {}, {}, {}};
  int cur = 0;
  for (int s = 0; s < nsteps; ++s) {
#if DMHA_HAS_TDM
    if (threadIdx.x < 32) __builtin_amdgcn_s_wait_tensorcnt(0);
#endif
    __syncthreads();  // buffer `cur` ready; prior reads of `cur^1` done
    if (s + 1 < nsteps) {
      const int kn = (s + 1) * 32;
#if DMHA_HAS_TDM
      if (threadIdx.x < 32) {
        tdm_load_2d(lds_off(&sA[cur ^ 1][0]), aG + kn, 32, A2R, S);
        tdm_load_2d(lds_off(&sB[cur ^ 1][0]), bG + kn, 32, A2C, S);
      }
#else
      for (int t = threadIdx.x; t < A2R * 32; t += 256)
        sA[cur ^ 1][t] = aG[(long)(t >> 5) * S + (t & 31) + kn];
      for (int t = threadIdx.x; t < A2C * 32; t += 256)
        sB[cur ^ 1][t] = bG[(long)(t >> 5) * S + (t & 31) + kn];
#endif
    }
    const v16h a = load_frag(&sA[cur][wave * 16 * 32], 32, 0);
#pragma unroll
    for (int nt = 0; nt < 4; ++nt) {
      const v16h b = load_frag(&sB[cur][nt * 16 * 32], 32, 0);
      acc[nt] = wmma16(a, b, acc[nt]);
    }
    cur ^= 1;
    __syncthreads();  // all reads of this buffer done before it is re-filled
  }

  const float* qk = QK + (long)bh * S * S;
  const float* Cb = Crow + (long)bh * S;
#pragma unroll
  for (int nt = 0; nt < 4; ++nt) {
#pragma unroll
    for (int r = 0; r < 8; ++r) {
      const int i = i0 + wave * 16 + acc_row(r);
      const int kg = c0 + nt * 16 + acc_col();
      float val = 0.f;
      if (kg <= i) {
        const float a1 = __expf(qk[(long)i * S + kg] - Cb[i]);
        val = 2.f * a1 - acc[nt][r];
      }
      outb[(long)i * S + kg] = (f16)val;
    }
  }
}

// ---------------------------------------------------------------------------
// 8) Row sums of A
// ---------------------------------------------------------------------------
__global__ __launch_bounds__(256) void rowsum(const f16* __restrict__ A16,
                                              float* __restrict__ rs) {
  const int row = blockIdx.x;  // bh*S + i
  const f16* a = A16 + (long)row * S;
  float s = 0.f;
  for (int j = threadIdx.x; j < S; j += blockDim.x) s += (float)a[j];
  __shared__ float red[256];
  red[threadIdx.x] = s;
  __syncthreads();
  for (int t = 128; t > 0; t >>= 1) {
    if ((int)threadIdx.x < t) red[threadIdx.x] += red[threadIdx.x + t];
    __syncthreads();
  }
  if (threadIdx.x == 0) rs[row] = red[0];
}

// ---------------------------------------------------------------------------
// 9) ctx = (A @ v) / rowsum   -> f32 [B,S,512]
// ---------------------------------------------------------------------------
__global__ __launch_bounds__(256) void av_gemm(const f16* __restrict__ A16,
                                               const f16* __restrict__ vT16,
                                               const float* __restrict__ rs,
                                               float* __restrict__ ctx) {
  const int wave = threadIdx.x >> 5;
  const int it = blockIdx.x * 8 + wave;  // 0..63
  const int bh = blockIdx.y;
  const f16* abase = A16 + (long)bh * S * S + (long)(it * 16) * S;
  const f16* vb = vT16 + (long)bh * Dh * S;
  v8f acc[4] = {{}, {}, {}, {}};
  const int kend = (it + 1) * 16;  // A zero beyond causal boundary
  for (int k0 = 0; k0 < kend; k0 += 32) {
    v16h a = load_frag(abase, S, k0);
#pragma unroll
    for (int nt2 = 0; nt2 < 4; ++nt2) {
      v16h b = load_frag(vb + (long)nt2 * 16 * S, S, k0);
      acc[nt2] = wmma16(a, b, acc[nt2]);
    }
  }
  const int b_ = bh >> 3, h = bh & 7;
#pragma unroll
  for (int nt2 = 0; nt2 < 4; ++nt2) {
#pragma unroll
    for (int r = 0; r < 8; ++r) {
      const int i = it * 16 + acc_row(r);
      const float inv = 1.f / rs[(long)bh * S + i];
      ctx[((long)(b_ * S + i)) * D + h * 64 + nt2 * 16 + acc_col()] =
          acc[nt2][r] * inv;
    }
  }
}

// ---------------------------------------------------------------------------
// 10) GroupNorm stats per (b, head-group) over (S, Dh)
// ---------------------------------------------------------------------------
__global__ __launch_bounds__(256) void gn_stats(const float* __restrict__ ctx,
                                                float* __restrict__ gnm,
                                                float* __restrict__ gnr) {
  const int bh = blockIdx.x;
  const int b_ = bh >> 3, h = bh & 7;
  float s = 0.f, ss = 0.f;
  for (int t = threadIdx.x; t < S * Dh; t += blockDim.x) {
    const int sp = t >> 6, d = t & 63;
    const float v = ctx[((long)(b_ * S + sp)) * D + h * 64 + d];
    s += v;
    ss += v * v;
  }
  __shared__ float r1[256], r2[256];
  r1[threadIdx.x] = s;
  r2[threadIdx.x] = ss;
  __syncthreads();
  for (int t = 128; t > 0; t >>= 1) {
    if ((int)threadIdx.x < t) {
      r1[threadIdx.x] += r1[threadIdx.x + t];
      r2[threadIdx.x] += r2[threadIdx.x + t];
    }
    __syncthreads();
  }
  if (threadIdx.x == 0) {
    const float inv_n = 1.f / (float)(S * Dh);
    const float mean = r1[0] * inv_n;
    const float var = r2[0] * inv_n - mean * mean;
    gnm[bh] = mean;
    gnr[bh] = rsqrtf(var + GN_EPS);
  }
}

// ---------------------------------------------------------------------------
// 11) GroupNorm apply -> f16 for the out-proj GEMM
// ---------------------------------------------------------------------------
__global__ __launch_bounds__(256) void gn_apply(const float* __restrict__ ctx,
                                                const float* __restrict__ gnm,
                                                const float* __restrict__ gnr,
                                                const float* __restrict__ gamma,
                                                const float* __restrict__ beta,
                                                f16* __restrict__ y16) {
  const long idx = (long)blockIdx.x * blockDim.x + threadIdx.x;
  if (idx >= (long)Bb * S * D) return;
  const int ch = (int)idx & 511;
  const int b_ = (int)(idx >> 19);
  const int bh = b_ * 8 + (ch >> 6);
  const float v = (ctx[idx] - gnm[bh]) * gnr[bh] * gamma[ch] + beta[ch];
  y16[idx] = (f16)v;
}

// ---------------------------------------------------------------------------
// 12) Output projection: [4096,512] x [512,512]^T + bias -> f32 d_out
// ---------------------------------------------------------------------------
__global__ __launch_bounds__(256) void out_gemm(const f16* __restrict__ y16,
                                                const f16* __restrict__ wo16,
                                                const float* __restrict__ ob,
                                                float* __restrict__ out) {
  const int wave = threadIdx.x >> 5;
  const int mt = blockIdx.x;             // 0..255
  const int nt = blockIdx.y * 8 + wave;  // 0..31
  const f16* abase = y16 + (long)mt * 16 * D;
  const f16* bbase = wo16 + (long)nt * 16 * D;
  v8f acc = {};
  for (int k0 = 0; k0 < D; k0 += 32) {
    v16h a = load_frag(abase, D, k0);
    v16h b = load_frag(bbase, D, k0);
    acc = wmma16(a, b, acc);
  }
  const int n = nt * 16 + acc_col();
  const float bias = ob[n];
#pragma unroll
  for (int r = 0; r < 8; ++r) {
    const int row = mt * 16 + acc_row(r);
    out[(long)row * D + n] = acc[r] + bias;
  }
}

}  // namespace dmha

// ---------------------------------------------------------------------------
extern "C" void kernel_launch(void* const* d_in, const int* in_sizes, int n_in,
                              void* d_out, int out_size, void* d_ws,
                              size_t ws_size, hipStream_t stream) {
  using namespace dmha;
  (void)in_sizes; (void)n_in; (void)out_size; (void)ws_size;

  const float* x = (const float*)d_in[0];       // [4,1024,512]
  const float* wproj = (const float*)d_in[1];   // [1536,512]
  const float* wproj_b = (const float*)d_in[2]; // [1536]
  const float* wout = (const float*)d_in[3];    // [512,512]
  const float* wout_b = (const float*)d_in[4];  // [512]
  const float* gamma = (const float*)d_in[5];   // [512]
  const float* beta = (const float*)d_in[6];    // [512]
  float* out = (float*)d_out;                   // [4,1024,512]

  char* p = (char*)d_ws;
  auto alloc = [&](size_t bytes) {
    char* r = p;
    p += (bytes + 255) & ~(size_t)255;
    return r;
  };
  f16* x16  = (f16*)alloc((size_t)BS * D * 2);
  f16* wp16 = (f16*)alloc((size_t)D3 * D * 2);
  f16* wo16 = (f16*)alloc((size_t)D * D * 2);
  f16* q16  = (f16*)alloc((size_t)BH * S * Dh * 2);
  f16* k16  = (f16*)alloc((size_t)BH * S * Dh * 2);
  f16* vT16 = (f16*)alloc((size_t)BH * Dh * S * 2);
  float* QK = (float*)alloc((size_t)BH * S * S * 4);   // 128 MB
  f16* E1   = (f16*)alloc((size_t)BH * S * S * 2);     // 64 MB
  f16* E2T  = (f16*)alloc((size_t)BH * S * S * 2);     // 64 MB
  f16* A16  = (f16*)alloc((size_t)BH * S * S * 2);     // 64 MB
  float* Mrow = (float*)alloc((size_t)BH * S * 4);
  float* Crow = (float*)alloc((size_t)BH * S * 4);
  float* rs   = (float*)alloc((size_t)BH * S * 4);
  float* ctx  = (float*)alloc((size_t)BS * D * 4);
  f16* y16    = (f16*)alloc((size_t)BS * D * 2);
  float* gnm  = (float*)alloc(256);
  float* gnr  = (float*)alloc(256);

  (void)hipMemsetAsync(Mrow, 0, (size_t)BH * S * 4, stream);

  {
    int n = BS * D;
    cvt_f32_f16<<<(n + 255) / 256, 256, 0, stream>>>(x, x16, n);
    n = D3 * D;
    cvt_f32_f16<<<(n + 255) / 256, 256, 0, stream>>>(wproj, wp16, n);
    n = D * D;
    cvt_f32_f16<<<(n + 255) / 256, 256, 0, stream>>>(wout, wo16, n);
  }
  qkv_gemm<<<dim3(BS / 16, D3 / 16 / 8), 256, 0, stream>>>(x16, wp16, wproj_b,
                                                           q16, k16, vT16);
  kk_rowmax<<<dim3(S / 16, S / 16 / 8, BH), 256, 0, stream>>>(k16, Mrow);
  qk_gemm<<<dim3(S / 16, S / 16 / 8, BH), 256, 0, stream>>>(q16, k16, QK);
  row_c<<<BH * S, 256, 0, stream>>>(QK, Mrow, Crow);
  {
    long n = (long)BH * S * S;
    e1_kernel<<<(unsigned)(n / 256), 256, 0, stream>>>(QK, Crow, Mrow, E1);
  }
  kk_e2<<<dim3(S / 16, S / 16 / 8, BH), 256, 0, stream>>>(k16, Mrow, E2T);
  // big einsum: 128x64 macro-tiles, TDM-staged LDS double buffering
  a2_gemm<<<dim3(S / A2R, S / A2C, BH), 256, 0, stream>>>(E1, E2T, QK, Crow,
                                                          A16);
  rowsum<<<BH * S, 256, 0, stream>>>(A16, rs);
  av_gemm<<<dim3(S / 16 / 8, BH), 256, 0, stream>>>(A16, vT16, rs, ctx);
  gn_stats<<<BH, 256, 0, stream>>>(ctx, gnm, gnr);
  {
    long n = (long)BS * D;
    gn_apply<<<(unsigned)((n + 255) / 256), 256, 0, stream>>>(ctx, gnm, gnr,
                                                              gamma, beta, y16);
  }
  out_gemm<<<dim3(BS / 16, D / 16 / 8), 256, 0, stream>>>(y16, wo16, wout_b,
                                                          out);
}
